// LightweightTemporalAttention_79070347919973
// MI455X (gfx1250) — compile-verified, matrix operand order checked by reading, not run on hardware
//
#include <hip/hip_runtime.h>
#include <math.h>

// Problem constants (from reference): x is (B, C, T, V), d_k = C/4
#define B_ 8
#define C_ 256
#define T_ 2048
#define V_ 25
#define D_ 64

typedef __attribute__((ext_vector_type(16))) __bf16        v16bf;
typedef __attribute__((ext_vector_type(8)))  float         v8f;
typedef __attribute__((ext_vector_type(4)))  float         v4f;
typedef __attribute__((ext_vector_type(4)))  unsigned int  v4u;

union V16U { v16bf v; v4u q[2]; };

__device__ __forceinline__ unsigned short f2bf(float f) {
  unsigned int u = __builtin_bit_cast(unsigned int, f);
  u += 0x7FFFu + ((u >> 16) & 1u);            // round-to-nearest-even
  return (unsigned short)(u >> 16);
}
__device__ __forceinline__ float bf2f(unsigned short h) {
  unsigned int u = ((unsigned int)h) << 16;
  return __builtin_bit_cast(float, u);
}

// ---------------------------------------------------------------------------
// Wave32 NT GEMM strip: C(16 x NT*16, f32) += A(16xK) * Bt(NT*16 x K)^T.
// A fragment loaded once per K=32 step, reused across NT WMMAs. All operand
// loads of a K-step are issued as one clause BEFORE the WMMA burst so the
// scheduler can overlap loads of step k+1 with the 4 WMMAs of step k.
// Per-lane loads follow the CDNA5 16-bit A/B VGPR striping:
//   lane group 0 (0-15) holds K = k..k+7 and k+16..k+23,
//   lane group 1 (16-31) holds K = k+8..k+15 and k+24..k+31.
// OM: 0 = f32 store, 1 = bf16 store, 2 = bf16 transposed store.
// ---------------------------------------------------------------------------
template <int NT, int OM>
__device__ __forceinline__ void wave_gemm_nt(const unsigned short* __restrict__ A, int lda,
                                             const unsigned short* __restrict__ Bt, int ldb,
                                             int K, float scale, void* __restrict__ Cout, int ldc) {
  const int lane = threadIdx.x & 31;
  const int r    = lane & 15;              // A row / B column within tile
  const int kh   = (lane >> 4) << 3;       // 0 or 8 (K-half select)
  const v4u* ap = (const v4u*)(A + (size_t)r * lda + kh);
  v8f acc[NT] = {};
  for (int k = 0; k < K; k += 32) {
    V16U a, b[NT];
    a.q[0] = ap[(k >> 3) + 0];             // K = k+kh .. k+kh+7
    a.q[1] = ap[(k >> 3) + 2];             // K = k+kh+16 .. k+kh+23
#pragma unroll
    for (int i = 0; i < NT; ++i) {
      const v4u* bp = (const v4u*)(Bt + (size_t)(i * 16 + r) * ldb + kh);
      b[i].q[0] = bp[(k >> 3) + 0];
      b[i].q[1] = bp[(k >> 3) + 2];
    }
#pragma unroll
    for (int i = 0; i < NT; ++i) {
      acc[i] = __builtin_amdgcn_wmma_f32_16x16x32_bf16(
          /*neg_a=*/false, a.v, /*neg_b=*/false, b[i].v,
          /*c_mod=*/(short)0, acc[i], /*reuse_a=*/false, /*reuse_b=*/false);
    }
  }
  // C/D layout: VGPR g, lanes 0-15 -> M=g, lanes 16-31 -> M=8+g; N = lane%16.
  const int n  = lane & 15;
  const int mb = (lane >> 4) << 3;
#pragma unroll
  for (int i = 0; i < NT; ++i) {
#pragma unroll
    for (int g = 0; g < 8; ++g) {
      const float val = acc[i][g] * scale;
      const int m  = mb + g;
      const int nn = i * 16 + n;
      if (OM == 0)      ((float*)Cout)[(size_t)m * ldc + nn] = val;
      else if (OM == 1) ((unsigned short*)Cout)[(size_t)m * ldc + nn] = f2bf(val);
      else              ((unsigned short*)Cout)[(size_t)nn * ldc + m] = f2bf(val);
    }
  }
}

// Wave id forced into an SGPR: scalar tile addressing + s_cbranch (no EXEC
// masking around WMMA, which requires EXEC = all ones).
__device__ __forceinline__ int uniform_wave_id() {
  return __builtin_amdgcn_readfirstlane((blockIdx.x * 256 + threadIdx.x) >> 5);
}

// ---- K0: convert Wq,Wk,Wv,Wp (f32) -> bf16 scratch [Wq|Wk|Wv|Wp] ----------
__global__ void lta_wconv(const float* __restrict__ Wq, const float* __restrict__ Wk,
                          const float* __restrict__ Wv, const float* __restrict__ Wp,
                          unsigned short* __restrict__ wbf) {
  const int i   = blockIdx.x * 256 + threadIdx.x;      // 4 * 16384
  const int sel = i >> 14;
  const int j   = i & 16383;
  const float* src = (sel == 0) ? Wq : (sel == 1) ? Wk : (sel == 2) ? Wv : Wp;
  wbf[i] = f2bf(src[j]);
}

// ---- K1: joint-mean pool over V, transpose to xt[b][t][c] (bf16) ----------
// Dominant HBM pass #1 (420 MB). Block = one (b, c, 256-t chunk): the source
// region is 6400 contiguous, 16B-aligned floats -> coalesced b128 NT loads
// staged through LDS; per-thread stride-25 LDS reads are bank-conflict-free
// (gcd(25,64)=1).
__global__ __launch_bounds__(256) void lta_pool(const float* __restrict__ x,
                                                unsigned short* __restrict__ xt) {
  __shared__ float buf[256 * V_];                      // 25600 B
  const int tid = threadIdx.x;
  const int tc  = blockIdx.x % (T_ / 256);
  const int c   = (blockIdx.x / (T_ / 256)) % C_;
  const int b   = blockIdx.x / ((T_ / 256) * C_);
  const v4f* s4 = (const v4f*)(x + ((size_t)(b * C_ + c) * T_ + tc * 256) * V_);
  v4f* d4 = (v4f*)buf;
  for (int i = tid; i < 256 * V_ / 4; i += 256)
    d4[i] = __builtin_nontemporal_load(s4 + i);
  __syncthreads();
  float s = 0.0f;
#pragma unroll
  for (int v = 0; v < V_; ++v) s += buf[tid * V_ + v];
  xt[((size_t)b * T_ + tc * 256 + tid) * C_ + c] = f2bf(s * (1.0f / V_));
}

// ---- K2: q,k,v projections: one wave = 16 rows x all 64 cols --------------
// q,k stored [b][t][d]; v stored transposed [b][d][t] so the PV GEMM is NT.
__global__ __launch_bounds__(256) void lta_qkv(const unsigned short* __restrict__ xt,
                                               const unsigned short* __restrict__ wbf,
                                               unsigned short* __restrict__ q,
                                               unsigned short* __restrict__ k,
                                               unsigned short* __restrict__ vT) {
  const int w     = uniform_wave_id();                       // 3*8*128 = 3072 waves
  const int which = w / (B_ * (T_ / 16));
  const int r     = w % (B_ * (T_ / 16));
  const int b     = r / (T_ / 16);
  const int mt    = r % (T_ / 16);
  const unsigned short* A  = xt + ((size_t)b * T_ + mt * 16) * C_;
  const unsigned short* Bt = wbf + (size_t)which * (D_ * C_);
  if (which == 2) {
    unsigned short* out = vT + (size_t)b * D_ * T_ + mt * 16;
    wave_gemm_nt<4, 2>(A, C_, Bt, C_, C_, 1.0f, out, T_);
  } else {
    unsigned short* out = (which == 0 ? q : k) + ((size_t)b * T_ + mt * 16) * D_;
    wave_gemm_nt<4, 1>(A, C_, Bt, C_, C_, 1.0f, out, D_);
  }
}

// ---- K3: S = (Q K^T) / sqrt(d_k), bf16 output (64 MB -> lives in 192MB L2)
__global__ __launch_bounds__(256) void lta_scores(const unsigned short* __restrict__ q,
                                                  const unsigned short* __restrict__ k,
                                                  unsigned short* __restrict__ S) {
  const int w  = uniform_wave_id();                          // 8*128*32 = 32768 waves
  const int b  = w / (128 * 32);
  const int rr = w % (128 * 32);
  const int mt = rr / 32;
  const int g4 = rr % 32;                                    // group of 4 N-tiles
  const unsigned short* A  = q + ((size_t)b * T_ + mt * 16) * D_;
  const unsigned short* Bt = k + ((size_t)b * T_ + g4 * 64) * D_;
  unsigned short* out = S + (size_t)b * T_ * T_ + (size_t)(mt * 16) * T_ + g4 * 64;
  wave_gemm_nt<4, 1>(A, D_, Bt, D_, D_, 0.125f, out, T_);
}

// ---- K4: row softmax over T, in place on bf16 S ---------------------------
__global__ __launch_bounds__(256) void lta_softmax(unsigned short* __restrict__ S) {
  __shared__ float red[256];
  const int tid = threadIdx.x;
  unsigned short* p = S + (size_t)blockIdx.x * T_;           // 16384 rows
  float v[8];
  float m = -1e30f;
#pragma unroll
  for (int j = 0; j < 8; ++j) { v[j] = bf2f(p[tid + 256 * j]); m = fmaxf(m, v[j]); }
  red[tid] = m; __syncthreads();
  for (int s = 128; s > 0; s >>= 1) {
    if (tid < s) red[tid] = fmaxf(red[tid], red[tid + s]);
    __syncthreads();
  }
  m = red[0]; __syncthreads();
  float sum = 0.0f;
#pragma unroll
  for (int j = 0; j < 8; ++j) { v[j] = __expf(v[j] - m); sum += v[j]; }
  red[tid] = sum; __syncthreads();
  for (int s = 128; s > 0; s >>= 1) {
    if (tid < s) red[tid] += red[tid + s];
    __syncthreads();
  }
  const float inv = 1.0f / red[0];
#pragma unroll
  for (int j = 0; j < 8; ++j) p[tid + 256 * j] = f2bf(v[j] * inv);
}

// ---- K5: x_att = P @ V. One wave = 16 t-rows x all 64 d-cols: each P
// row-strip (16x2048) is streamed from L2 exactly once. ---------------------
__global__ __launch_bounds__(256) void lta_pv(const unsigned short* __restrict__ P,
                                              const unsigned short* __restrict__ vT,
                                              unsigned short* __restrict__ xatt) {
  const int w  = uniform_wave_id();                          // 8*128 = 1024 waves
  const int b  = w / 128;
  const int mt = w % 128;
  const unsigned short* A  = P  + (size_t)b * T_ * T_ + (size_t)(mt * 16) * T_;
  const unsigned short* Bt = vT + (size_t)b * D_ * T_;
  unsigned short* out = xatt + ((size_t)b * T_ + mt * 16) * D_;
  wave_gemm_nt<4, 1>(A, T_, Bt, T_, T_, 1.0f, out, D_);
}

// ---- K6: y = x_att @ Wp^T  (btd,cd->btc) ----------------------------------
__global__ __launch_bounds__(256) void lta_proj(const unsigned short* __restrict__ xatt,
                                                const unsigned short* __restrict__ wp,
                                                unsigned short* __restrict__ y) {
  const int w  = uniform_wave_id();                          // 8*128*4 = 4096 waves
  const int b  = w / (128 * 4);
  const int rr = w % (128 * 4);
  const int mt = rr / 4;
  const int cg = rr % 4;                                     // group of 4 N-tiles
  const unsigned short* A  = xatt + ((size_t)b * T_ + mt * 16) * D_;
  const unsigned short* Bt = wp + (size_t)(cg * 64) * D_;
  unsigned short* out = y + ((size_t)b * T_ + mt * 16) * C_ + cg * 64;
  wave_gemm_nt<4, 1>(A, D_, Bt, D_, D_, 1.0f, out, C_);
}

// ---- K7: out = x + sigmoid(gate) * y broadcast over V ---------------------
// Dominant HBM pass #2 (~840 MB). 4 elems/thread, non-temporal b128 in/out so
// this stream never thrashes L2.
__global__ __launch_bounds__(256) void lta_out(const float* __restrict__ x,
                                               const unsigned short* __restrict__ y,
                                               const float* __restrict__ gate,
                                               float* __restrict__ out) {
  const unsigned int id4 = ((unsigned int)blockIdx.x * 256 + threadIdx.x) * 4;
  const float g = 1.0f / (1.0f + __expf(-gate[0]));
  v4f xv = __builtin_nontemporal_load((const v4f*)(x + id4));
  v4f res;
#pragma unroll
  for (int j = 0; j < 4; ++j) {
    const unsigned int e  = id4 + j;              // flat (b,c,t,v), v fastest
    const unsigned int r  = e / V_;
    const unsigned int t  = r % T_;
    const unsigned int r2 = r / T_;
    const unsigned int c  = r2 % C_;
    const unsigned int b  = r2 / C_;
    res[j] = xv[j] + g * bf2f(y[((size_t)b * T_ + t) * C_ + c]);
  }
  __builtin_nontemporal_store(res, (v4f*)(out + id4));
}

extern "C" void kernel_launch(void* const* d_in, const int* in_sizes, int n_in,
                              void* d_out, int out_size, void* d_ws, size_t ws_size,
                              hipStream_t stream) {
  (void)in_sizes; (void)n_in; (void)out_size; (void)ws_size;
  const float* x    = (const float*)d_in[0];
  const float* Wq   = (const float*)d_in[1];
  const float* Wk   = (const float*)d_in[2];
  const float* Wv   = (const float*)d_in[3];
  const float* Wp   = (const float*)d_in[4];
  const float* gate = (const float*)d_in[5];
  float* out = (float*)d_out;

  char* ws = (char*)d_ws;
  unsigned short* wbf  = (unsigned short*)(ws + 0);         //  4*64*256 bf16 = 128 KB
  unsigned short* xt   = (unsigned short*)(ws + 131072);    //  B*T*C  bf16  =   8 MB
  unsigned short* q    = (unsigned short*)(ws + 8519680);   //  B*T*D  bf16  =   2 MB
  unsigned short* k    = (unsigned short*)(ws + 10616832);  //  B*T*D  bf16  =   2 MB
  unsigned short* vT   = (unsigned short*)(ws + 12713984);  //  B*D*T  bf16  =   2 MB
  unsigned short* S    = (unsigned short*)(ws + 14811136);  //  B*T*T  bf16  =  64 MB
  unsigned short* xatt = (unsigned short*)(ws + 81920000);  //  B*T*D  bf16  =   2 MB
  unsigned short* y    = (unsigned short*)(ws + 84017152);  //  B*T*C  bf16  =   8 MB
  // total scratch: ~92.4 MB

  lta_wconv  <<<256,    256, 0, stream>>>(Wq, Wk, Wv, Wp, wbf);
  lta_pool   <<<16384,  256, 0, stream>>>(x, xt);                 // B*C*(T/256)
  lta_qkv    <<<384,    256, 0, stream>>>(xt, wbf, q, k, vT);     // 3072 waves
  lta_scores <<<4096,   256, 0, stream>>>(q, k, S);               // 32768 waves
  lta_softmax<<<16384,  256, 0, stream>>>(S);                     // B*T rows
  lta_pv     <<<128,    256, 0, stream>>>(S, vT, xatt);           // 1024 waves
  lta_proj   <<<512,    256, 0, stream>>>(xatt, wbf + 3 * 16384, y); // 4096 waves
  lta_out    <<<102400, 256, 0, stream>>>(x, y, gate, out);       // B*C*T*V / 1024
}
